// CSWM_21406117003665
// MI455X (gfx1250) — matrix operand order repaired and checked by reading, not compiled, not run on hardware
//
#include <hip/hip_runtime.h>
#include <hip/hip_bf16.h>
#include <cstdint>
#include <cstddef>

// ---------------------------------------------------------------------------
// C-SWM transition GNN on MI455X (gfx1250, wave32, WMMA).
//   B=512, K=16 nodes, D=512, H=1024, A=20.  Edge quirk: k = K-1 = 15 nodes,
//   E = 512*15*14 = 107520 edges, endpoints index flat rows 0..7679.
//
// Heavy math: v_wmma_f32_16x16x32_f16 tiled GEMMs.
//   - generic GEMM: A *and* B panels double-buffered in LDS, filled with
//     global_load_async_to_lds_b128 (ASYNCcnt) -> one copy per WG per k-step
//     (kills the 8x cross-wave B redundancy); inner loop is ds_load-fed WMMA.
//   - LDS rows padded to 80B stride (bank stride 20 -> conflict-free frags).
//   - LN GEMM: full-row workgroups, fused bias+LayerNorm+ReLU epilogue,
//     2-deep software pipeline on (disjoint) per-wave B fragments.
// ---------------------------------------------------------------------------

typedef __attribute__((ext_vector_type(16))) _Float16 v16h;
typedef __attribute__((ext_vector_type(8)))  _Float16 h8;   // 16 bytes
typedef __attribute__((ext_vector_type(4)))  _Float16 h4;   // 8 bytes
typedef __attribute__((ext_vector_type(2)))  _Float16 h2;   // 4 bytes
typedef __attribute__((ext_vector_type(8)))  float    v8f;
typedef __attribute__((ext_vector_type(4)))  float    f4;

union AFrag { v16h v; h8 h[2]; };

__device__ __forceinline__ void ldB(AFrag& f, const _Float16* p)
{
    f.h[0] = *(const h8*)p;
    f.h[1] = *(const h8*)(p + 8);
}

#define LDSW 40   // padded LDS row stride in halves (80B: 16B-aligned, bank stride 20)

// ---------------------------------------------------------------------------
// Generic tiled WMMA GEMM:  C[M,N] = A[M,K] * B[K,N] (+ bias*bscale)(+resid)
//   A: fp16 row-major.  BT: fp16 transposed [N][K].
//   Block tile 128x128, 256 threads (8 wave32s): wave w owns M-tile w and all
//   8 N-tiles.  A and B panels staged in LDS via async-to-LDS, double-buffered
//   (~64 flop per unique L2 byte, B loaded once per WG per k-step).
//   Requires M%128==0, N%128==0, K%32==0 (true at all call sites).
// ---------------------------------------------------------------------------
template<bool OUT_HALF, bool RELU, bool ACC>
__global__ __launch_bounds__(256)
void wmma_gemm(const _Float16* __restrict__ A, const _Float16* __restrict__ BT,
               void* __restrict__ Cptr, const float* __restrict__ bias, float bscale,
               const float* __restrict__ resid, int M, int N, int K)
{
    __shared__ _Float16 As[2][128 * LDSW];   // [row][k]
    __shared__ _Float16 Bs[2][128 * LDSW];   // [col][k]
    const int tid = threadIdx.x;
    const int w   = tid >> 5;          // wave = M-tile 0..7
    const int L   = tid & 31;
    const int ln  = L & 15;
    const bool hi = (L >= 16);
    const int m0  = blockIdx.y * 128;
    const int n0  = blockIdx.x * 128;

    v8f acc[8];
#pragma unroll
    for (int t = 0; t < 8; ++t) acc[t] = v8f{};

    // Panel loaders: 128 rows(cols) x 32 halves, 16 halves (2x16B) per thread.
    const int    prow  = tid >> 1;          // 0..127
    const int    pko   = (tid & 1) * 16;    // 0 or 16
    const size_t abase = (size_t)(m0 + prow) * (size_t)K + pko;
    const _Float16* bsrc = BT + (size_t)(n0 + prow) * (size_t)K + pko;
    __builtin_prefetch(bsrc, 0, 1);          // global_prefetch_b8 (warm L2)

    // synchronous preload of k-step 0 panels
    *(h8*)&As[0][prow * LDSW + pko]     = *(const h8*)&A[abase];
    *(h8*)&As[0][prow * LDSW + pko + 8] = *(const h8*)&A[abase + 8];
    *(h8*)&Bs[0][prow * LDSW + pko]     = *(const h8*)&bsrc[0];
    *(h8*)&Bs[0][prow * LDSW + pko + 8] = *(const h8*)&bsrc[8];
    __syncthreads();

    int p = 0;
    for (int k0 = 0; k0 < K; k0 += 32) {
        const bool more = (k0 + 32) < K;

        // async-stage next A and B panels into the other LDS buffers
        // (ASYNCcnt path; overlaps the whole WMMA burst below)
        if (more) {
            const uint32_t da = (uint32_t)(uintptr_t)&As[p ^ 1][prow * LDSW + pko];
            const _Float16* ga = &A[abase + k0 + 32];
            asm volatile("global_load_async_to_lds_b128 %0, %1, off"
                         :: "v"(da), "v"(ga) : "memory");
            asm volatile("global_load_async_to_lds_b128 %0, %1, off"
                         :: "v"(da + 16u), "v"(ga + 8) : "memory");
            const uint32_t db = (uint32_t)(uintptr_t)&Bs[p ^ 1][prow * LDSW + pko];
            const _Float16* gb = bsrc + k0 + 32;
            asm volatile("global_load_async_to_lds_b128 %0, %1, off"
                         :: "v"(db), "v"(gb) : "memory");
            asm volatile("global_load_async_to_lds_b128 %0, %1, off"
                         :: "v"(db + 16u), "v"(gb + 8) : "memory");
        }

        // A fragment (contiguous k order: lanes 0-15 K0-15,
        // lanes 16-31 K16-31, matching B)
        AFrag af;
        const int ar = (w * 16 + ln) * LDSW + (hi ? 16 : 0);
        af.h[0] = *(const h8*)&As[p][ar];
        af.h[1] = *(const h8*)&As[p][ar + 8];

        // B fragments from LDS (lanes 0-15 K0-15 @ col, lanes 16-31 K16-31)
#pragma unroll
        for (int t = 0; t < 8; ++t) {
            AFrag bf;
            const int bo = (t * 16 + ln) * LDSW + (hi ? 16 : 0);
            bf.h[0] = *(const h8*)&Bs[p][bo];
            bf.h[1] = *(const h8*)&Bs[p][bo + 8];
            acc[t] = __builtin_amdgcn_wmma_f32_16x16x32_f16(
                         false, af.v, false, bf.v, (short)0, acc[t], false, false);
        }

        if (more) {
            asm volatile("s_wait_asynccnt 0x0" ::: "memory");
            __syncthreads();
            p ^= 1;
        }
    }

    // Epilogue.  C layout: lane ln = column, VGPR r -> row r (lanes 0-15) /
    // r+8 (lanes 16-31).
#pragma unroll
    for (int t = 0; t < 8; ++t) {
        const int col = n0 + t * 16 + ln;
        const float bv = bias ? bias[col] * bscale : 0.0f;
#pragma unroll
        for (int r = 0; r < 8; ++r) {
            const int row = m0 + w * 16 + r + (hi ? 8 : 0);
            const size_t o = (size_t)row * (size_t)N + col;
            float v = acc[t][r] + bv;
            if (resid) v += resid[o];
            if (RELU)  v = v > 0.f ? v : 0.f;
            if (OUT_HALF) {
                ((_Float16*)Cptr)[o] = (_Float16)v;
            } else {
                float* C = (float*)Cptr;
                if (ACC) v += C[o];
                C[o] = v;
            }
        }
    }
}

// ---------------------------------------------------------------------------
// WMMA GEMM with fused bias + LayerNorm + ReLU epilogue (N fixed = 1024).
//   512 threads (16 wave32s), block tile 32x1024: wave w -> M-tile (w&1),
//   N-group (w>>1) of 8 tiles.  Each WG owns full rows -> LN stats local,
//   reduced across waves via LDS.  Per-wave B columns are disjoint, loaded
//   straight from L2-resident transposed weights with a 2-deep pipeline.
//   Output fp16.  Requires M%32==0, K%32==0.
// ---------------------------------------------------------------------------
__global__ __launch_bounds__(512)
void wmma_gemm_ln(const _Float16* __restrict__ A, const _Float16* __restrict__ BT,
                  const float* __restrict__ bias, const float* __restrict__ gamma,
                  const float* __restrict__ beta, _Float16* __restrict__ Y, int K)
{
    constexpr int N = 1024;
    __shared__ _Float16 As[2][32 * LDSW];
    __shared__ float red_s[32][128];
    __shared__ float red_q[32][128];
    __shared__ float mu_s[32];
    __shared__ float is_s[32];

    const int tid = threadIdx.x;
    const int w   = tid >> 5;          // 0..15
    const int L   = tid & 31;
    const int ln  = L & 15;
    const bool hi = (L >= 16);
    const int mt  = w & 1;             // M sub-tile 0..1
    const int ng  = w >> 1;            // N group 0..7 (8 tiles each)
    const int m0  = blockIdx.x * 32;

    v8f acc[8];
#pragma unroll
    for (int t = 0; t < 8; ++t) acc[t] = v8f{};

    // A tile loader: 32x32 halves = 1024, 2 halves (4B) per thread.
    const int    arow  = tid >> 4;
    const int    acol  = (tid & 15) * 2;
    const size_t abase = (size_t)(m0 + arow) * (size_t)K + acol;

    const _Float16* bbase[8];
#pragma unroll
    for (int t = 0; t < 8; ++t) {
        bbase[t] = BT + (size_t)((ng * 8 + t) * 16 + ln) * (size_t)K + (hi ? 16 : 0);
        __builtin_prefetch(bbase[t], 0, 1);
    }

    *(h2*)&As[0][arow * LDSW + acol] = *(const h2*)&A[abase];
    __syncthreads();

    AFrag pre0, pre1;
    ldB(pre0, bbase[0]);
    ldB(pre1, bbase[1]);

    int p = 0;
    for (int k0 = 0; k0 < K; k0 += 32) {
        const bool more = (k0 + 32) < K;

        if (more) {
            const uint32_t d0 = (uint32_t)(uintptr_t)&As[p ^ 1][arow * LDSW + acol];
            const _Float16* g0 = &A[abase + k0 + 32];
            asm volatile("global_load_async_to_lds_b32 %0, %1, off"
                         :: "v"(d0), "v"(g0) : "memory");
        }

        AFrag af;
        const int ar = (mt * 16 + ln) * LDSW + (hi ? 16 : 0);
        af.h[0] = *(const h8*)&As[p][ar];
        af.h[1] = *(const h8*)&As[p][ar + 8];

        // 2-deep B software pipeline: load tile t+2 between WMMAs; slots 6,7
        // preload the first two fragments of the next k-step.
        AFrag bf[8];
        bf[0] = pre0;
        bf[1] = pre1;
        const int kn = more ? k0 + 32 : k0;
#pragma unroll
        for (int t = 0; t < 8; ++t) {
            if (t < 6)       ldB(bf[t + 2], bbase[t + 2] + k0);
            else if (t == 6) ldB(pre0, bbase[0] + kn);
            else             ldB(pre1, bbase[1] + kn);
            acc[t] = __builtin_amdgcn_wmma_f32_16x16x32_f16(
                         false, af.v, false, bf[t].v, (short)0, acc[t], false, false);
        }

        if (more) {
            asm volatile("s_wait_asynccnt 0x0" ::: "memory");
            __syncthreads();
            p ^= 1;
        }
    }

    // bias + per-lane partial LN stats
#pragma unroll
    for (int r = 0; r < 8; ++r) {
        float s = 0.f, q = 0.f;
#pragma unroll
        for (int t = 0; t < 8; ++t) {
            const int col = (ng * 8 + t) * 16 + ln;
            float v = acc[t][r] + bias[col];
            acc[t][r] = v;
            s += v;
            q += v * v;
        }
        const int R = mt * 16 + r + (hi ? 8 : 0);
        red_s[R][ng * 16 + ln] = s;
        red_q[R][ng * 16 + ln] = q;
    }
    __syncthreads();

    if (tid < 32) {
        float s = 0.f, q = 0.f;
        for (int i = 0; i < 128; ++i) { s += red_s[tid][i]; q += red_q[tid][i]; }
        const float mu  = s * (1.0f / N);
        const float var = q * (1.0f / N) - mu * mu;   // biased var (jnp.var)
        mu_s[tid] = mu;
        is_s[tid] = rsqrtf(var + 1e-5f);
    }
    __syncthreads();

#pragma unroll
    for (int r = 0; r < 8; ++r) {
        const int R = mt * 16 + r + (hi ? 8 : 0);
        const float mu = mu_s[R], is = is_s[R];
#pragma unroll
        for (int t = 0; t < 8; ++t) {
            const int col = (ng * 8 + t) * 16 + ln;
            float y = (acc[t][r] - mu) * is * gamma[col] + beta[col];
            y = y > 0.f ? y : 0.f;
            Y[(size_t)(m0 + R) * N + col] = (_Float16)y;
        }
    }
}

// ---------------------------------------------------------------------------
// Pack / pointwise kernels
// ---------------------------------------------------------------------------
__global__ void f32_to_f16(const float* __restrict__ s, _Float16* __restrict__ d, long n)
{
    for (long i = blockIdx.x * 256L + threadIdx.x; i < n; i += (long)gridDim.x * 256)
        d[i] = (_Float16)s[i];
}

// wt[n*K+k] = w[(rowOff+k)*N + n]  (fp32 -> fp16, transpose)
__global__ void pack_wT(const float* __restrict__ w, _Float16* __restrict__ wt,
                        int rowOff, int K, int N, long total)
{
    for (long idx = blockIdx.x * 256L + threadIdx.x; idx < total;
         idx += (long)gridDim.x * 256) {
        const int k = (int)(idx / N), n = (int)(idx % N);
        wt[(size_t)n * K + k] = (_Float16)w[(size_t)(rowOff + k) * N + n];
    }
}

// e_w0 [1536][1024]: wpT = (W0a+W0b)^T, wcT = W0c^T  (both [1024][512] fp16)
__global__ void pack_w0T(const float* __restrict__ w, _Float16* __restrict__ wpT,
                         _Float16* __restrict__ wcT, long total /* 512*1024 */)
{
    for (long idx = blockIdx.x * 256L + threadIdx.x; idx < total;
         idx += (long)gridDim.x * 256) {
        const int k = (int)(idx / 1024), n = (int)(idx % 1024);
        wpT[(size_t)n * 512 + k] =
            (_Float16)(w[(size_t)k * 1024 + n] + w[(size_t)(k + 512) * 1024 + n]);
        wcT[(size_t)n * 512 + k] = (_Float16)w[(size_t)(k + 1024) * 1024 + n];
    }
}

// Edge layer-0 assemble: X0[e] = relu(P[row] + Q[col] + b0), fp16.
// Edge e (global): b=e/210, i=(e%210)/14, jj=e%14, j=jj+(jj>=i); row=b*15+i.
__global__ __launch_bounds__(256)
void edge_h0(const _Float16* __restrict__ P, const _Float16* __restrict__ Q,
             const float* __restrict__ b0, _Float16* __restrict__ X0, int e0)
{
    const int el = blockIdx.x;
    const int e  = e0 + el;
    const int b  = e / 210, r2 = e % 210;
    const int i  = r2 / 14, jj = r2 % 14;
    const int j  = jj + (jj >= i ? 1 : 0);
    const size_t rp = (size_t)(b * 15 + i) * 1024;
    const size_t cp = (size_t)(b * 15 + j) * 1024;
    const size_t o  = (size_t)el * 1024;
    const int c0 = threadIdx.x * 4;
    const h4 pv = *(const h4*)&P[rp + c0];
    const h4 qv = *(const h4*)&Q[cp + c0];
    const f4 bv = *(const f4*)&b0[c0];
    h4 ov;
#pragma unroll
    for (int u = 0; u < 4; ++u) {
        float x = (float)pv[u] + (float)qv[u] + bv[u];
        ov[u] = (_Float16)(x > 0.f ? x : 0.f);
    }
    *(h4*)&X0[o + c0] = ov;
}

// Segment-sum of 14 consecutive edge rows (edges grouped by source node).
__global__ __launch_bounds__(256)
void reduce14(const _Float16* __restrict__ Y, _Float16* __restrict__ Ysum, int n0)
{
    const int nl = blockIdx.x;
    const int ng = n0 + nl;
    const int b  = nl / 15, i = nl % 15;
    const size_t base = (size_t)(b * 210 + i * 14) * 1024;
    const int c0 = threadIdx.x * 4;
    f4 s = f4{};
#pragma unroll
    for (int t = 0; t < 14; ++t) {
        const h4 y = *(const h4*)&Y[base + (size_t)t * 1024 + c0];
#pragma unroll
        for (int u = 0; u < 4; ++u) s[u] += (float)y[u];
    }
    h4 ov;
#pragma unroll
    for (int u = 0; u < 4; ++u) ov[u] = (_Float16)s[u];
    *(h4*)&Ysum[(size_t)ng * 1024 + c0] = ov;
}

// Node layer-0 assemble: X2[j] = relu(U[j] + n_b0 + one-hot action row), fp16.
__global__ __launch_bounds__(256)
void node_h0(const float* __restrict__ U, const float* __restrict__ nb0,
             const float* __restrict__ nw0, const int* __restrict__ action,
             _Float16* __restrict__ X2)
{
    const int j = blockIdx.x;
    const int b = j >> 4, k = j & 15;
    const int a = action[b];
    const int an = a / 20, ap = a % 20;
    const bool hit = (k == an);
    const float* wrow = nw0 + (size_t)(512 + ap) * 1024;  // action rows 512..531
    const size_t o = (size_t)j * 1024;
    const int c0 = threadIdx.x * 4;
    const f4 uv = *(const f4*)&U[o + c0];
    const f4 bv = *(const f4*)&nb0[c0];
    h4 ov;
#pragma unroll
    for (int u = 0; u < 4; ++u) {
        float v = uv[u] + bv[u] + (hit ? wrow[c0 + u] : 0.f);
        ov[u] = (_Float16)(v > 0.f ? v : 0.f);
    }
    *(h4*)&X2[o + c0] = ov;
}

// ---------------------------------------------------------------------------
extern "C" void kernel_launch(void* const* d_in, const int* in_sizes, int n_in,
                              void* d_out, int out_size, void* d_ws, size_t ws_size,
                              hipStream_t stream)
{
    (void)in_sizes; (void)n_in; (void)out_size; (void)ws_size;

    const float* states = (const float*)d_in[0];
    const int*   action = (const int*)d_in[1];
    const float* e_w0 = (const float*)d_in[2];
    const float* e_b0 = (const float*)d_in[3];
    const float* e_w1 = (const float*)d_in[4];
    const float* e_b1 = (const float*)d_in[5];
    const float* e_g  = (const float*)d_in[6];
    const float* e_be = (const float*)d_in[7];
    const float* e_w2 = (const float*)d_in[8];
    const float* e_b2 = (const float*)d_in[9];
    const float* n_w0 = (const float*)d_in[10];
    const float* n_b0 = (const float*)d_in[11];
    const float* n_w1 = (const float*)d_in[12];
    const float* n_b1 = (const float*)d_in[13];
    const float* n_g  = (const float*)d_in[14];
    const float* n_be = (const float*)d_in[15];
    const float* n_w2 = (const float*)d_in[16];
    const float* n_b2 = (const float*)d_in[17];
    float* out = (float*)d_out;

    constexpr int Nn = 8192;     // B*K flat nodes
    constexpr int Nr = 7680;     // B*15 edge-referenced nodes
    constexpr int CH = 4, BC = 512 / CH;           // 4 chunks of 128 batches
    constexpr int EC = BC * 210;                   // 26880 edges / chunk
    constexpr int NC = BC * 15;                    // 1920 nodes / chunk

    // workspace bump allocator (~261 MB total)
    char* ws = (char*)d_ws; size_t off = 0;
    auto alloc = [&](size_t bytes) -> void* {
        off = (off + 255) & ~(size_t)255;
        void* p = ws + off; off += bytes; return p;
    };
    _Float16* s16    = (_Float16*)alloc((size_t)Nn * 512 * 2);
    _Float16* wpT    = (_Float16*)alloc((size_t)1024 * 512 * 2);
    _Float16* wcT    = (_Float16*)alloc((size_t)1024 * 512 * 2);
    _Float16* ew1T   = (_Float16*)alloc((size_t)1024 * 1024 * 2);
    _Float16* ew2T   = (_Float16*)alloc((size_t)1024 * 1024 * 2);
    _Float16* nw0aT  = (_Float16*)alloc((size_t)1024 * 512 * 2);
    _Float16* nw0cT  = (_Float16*)alloc((size_t)1024 * 1024 * 2);
    _Float16* nw1T   = (_Float16*)alloc((size_t)1024 * 1024 * 2);
    _Float16* nw2T   = (_Float16*)alloc((size_t)512 * 1024 * 2);
    _Float16* P16    = (_Float16*)alloc((size_t)Nr * 1024 * 2);
    _Float16* Q16    = (_Float16*)alloc((size_t)Nr * 1024 * 2);
    _Float16* X0c    = (_Float16*)alloc((size_t)EC * 1024 * 2);
    _Float16* Yc     = (_Float16*)alloc((size_t)EC * 1024 * 2);
    _Float16* Ysum   = (_Float16*)alloc((size_t)Nr * 1024 * 2);
    _Float16* agg16  = (_Float16*)alloc((size_t)Nr * 1024 * 2);
    float*    U      = (float*)   alloc((size_t)Nn * 1024 * 4);
    _Float16* X2     = (_Float16*)alloc((size_t)Nn * 1024 * 2);
    _Float16* Z      = (_Float16*)alloc((size_t)Nn * 1024 * 2);

    const dim3 blk(256);
    const dim3 blkLN(512);

    // ---- pack: states + weights to fp16 (weights transposed [N][K]) -------
    f32_to_f16<<<4096, blk, 0, stream>>>(states, s16, (long)Nn * 512);
    pack_w0T  <<<2048, blk, 0, stream>>>(e_w0, wpT, wcT, 512L * 1024);
    pack_wT   <<<4096, blk, 0, stream>>>(e_w1, ew1T, 0,   1024, 1024, 1024L * 1024);
    pack_wT   <<<4096, blk, 0, stream>>>(e_w2, ew2T, 0,   1024, 1024, 1024L * 1024);
    pack_wT   <<<2048, blk, 0, stream>>>(n_w0, nw0aT, 0,  512,  1024, 512L * 1024);
    pack_wT   <<<4096, blk, 0, stream>>>(n_w0, nw0cT, 532, 1024, 1024, 1024L * 1024);
    pack_wT   <<<4096, blk, 0, stream>>>(n_w1, nw1T, 0,   1024, 1024, 1024L * 1024);
    pack_wT   <<<2048, blk, 0, stream>>>(n_w2, nw2T, 0,   1024, 512,  1024L * 512);

    // ---- edge layer-0 factorized GEMMs: P = X(Wa+Wb), Q = X Wc ------------
    wmma_gemm<true, false, false><<<dim3(1024 / 128, Nr / 128), blk, 0, stream>>>(
        s16, wpT, P16, nullptr, 0.f, nullptr, Nr, 1024, 512);
    wmma_gemm<true, false, false><<<dim3(1024 / 128, Nr / 128), blk, 0, stream>>>(
        s16, wcT, Q16, nullptr, 0.f, nullptr, Nr, 1024, 512);

    // ---- per-chunk: assemble h0, fused GEMM+LN+ReLU, segment reduce -------
    for (int c = 0; c < CH; ++c) {
        edge_h0<<<EC, blk, 0, stream>>>(P16, Q16, e_b0, X0c, c * EC);
        wmma_gemm_ln<<<EC / 32, blkLN, 0, stream>>>(X0c, ew1T, e_b1, e_g, e_be, Yc, 1024);
        reduce14<<<NC, blk, 0, stream>>>(Yc, Ysum, c * NC);
    }

    // ---- agg = Ysum @ W2 + 14*b2 (segment-sum commuted past layer-2) ------
    wmma_gemm<true, false, false><<<dim3(1024 / 128, Nr / 128), blk, 0, stream>>>(
        Ysum, ew2T, agg16, e_b2, 14.0f, nullptr, Nr, 1024, 1024);

    // ---- node layer-0: U = X @ nw0a ; U += agg @ nw0c (rows < 7680) -------
    wmma_gemm<false, false, false><<<dim3(1024 / 128, Nn / 128), blk, 0, stream>>>(
        s16, nw0aT, U, nullptr, 0.f, nullptr, Nn, 1024, 512);
    wmma_gemm<false, false, true><<<dim3(1024 / 128, Nr / 128), blk, 0, stream>>>(
        agg16, nw0cT, U, nullptr, 0.f, nullptr, Nr, 1024, 1024);
    node_h0<<<Nn, blk, 0, stream>>>(U, n_b0, n_w0, action, X2);

    // ---- node layer-1 fused GEMM+LN+ReLU ----------------------------------
    wmma_gemm_ln<<<Nn / 32, blkLN, 0, stream>>>(X2, nw1T, n_b1, n_g, n_be, Z, 1024);

    // ---- node layer-2 + bias + residual -> fp32 output --------------------
    wmma_gemm<false, false, false><<<dim3(512 / 128, Nn / 128), blk, 0, stream>>>(
        Z, nw2T, out, n_b2, 1.0f, states, Nn, 512, 1024);
}
